// CrystalGraphAttention_89859305767778
// MI455X (gfx1250) — matrix operand order repaired and checked by reading, not compiled, not run on hardware
//
#include <hip/hip_runtime.h>
#include <math.h>

#define N_NODES 100000
#define E_EDGES 1600000
#define HID 128
#define QKV_COLS (3 * HID)

typedef __attribute__((ext_vector_type(16))) __bf16 v16bf;
typedef __attribute__((ext_vector_type(8)))  float  v8f;

union Frag { uint4 u[2]; v16bf v; };
union Acc  { v8f v; float f[8]; };

// ---------- monotone float<->uint key for atomicMax-based segment max ----------
__device__ __forceinline__ unsigned fkey(float f) {
    unsigned u = __float_as_uint(f);
    return (u & 0x80000000u) ? ~u : (u | 0x80000000u);
}
__device__ __forceinline__ float funkey(unsigned k) {
    return (k & 0x80000000u) ? __uint_as_float(k & 0x7fffffffu) : __uint_as_float(~k);
}

// ---------- zero fill ----------
__global__ void k_zero(float* p, int n) {
    int i = blockIdx.x * blockDim.x + threadIdx.x;
    if (i < n) p[i] = 0.f;
}

// ---------- hc = h + frac_coords @ frac_W + frac_b, cast to bf16 ----------
__global__ void k_hc(const float* __restrict__ h, const float* __restrict__ fc,
                     const float* __restrict__ fW, const float* __restrict__ fb,
                     __bf16* __restrict__ hc) {
    int i = blockIdx.x * blockDim.x + threadIdx.x;
    if (i >= N_NODES * HID) return;
    int node = i >> 7, j = i & 127;
    float x = h[i] + fb[j];
#pragma unroll
    for (int c = 0; c < 3; ++c) x += fc[node * 3 + c] * fW[c * HID + j];
    hc[i] = (__bf16)x;
}

// ---------- pack [qW|kW|vW] transposed (out-col major) into bf16 + bias concat ----------
__global__ void k_prep_qkv(const float* qW, const float* qb,
                           const float* kW, const float* kb,
                           const float* vW, const float* vb,
                           __bf16* __restrict__ wT, float* __restrict__ bc) {
    int i = blockIdx.x * blockDim.x + threadIdx.x;
    if (i >= QKV_COLS * HID) return;
    int o = i / HID, in = i % HID;
    const float* W = (o < HID) ? qW : (o < 2 * HID ? kW : vW);
    int oo = o & 127;
    wT[(size_t)o * HID + in] = (__bf16)W[in * HID + oo];
    if (in == 0) {
        const float* b = (o < HID) ? qb : (o < 2 * HID ? kb : vb);
        bc[o] = b[oo];
    }
}

__global__ void k_prep_o(const float* __restrict__ oW, __bf16* __restrict__ wT) {
    int i = blockIdx.x * blockDim.x + threadIdx.x;
    if (i >= HID * HID) return;
    int o = i / HID, in = i % HID;
    wT[(size_t)o * HID + in] = (__bf16)oW[in * HID + o];
}

// ---------- bf16 WMMA GEMM: C[M x 16*ctiles] = A[M x 128] * Wt^T + bias ----------
// A row-major (stride 128), Wt row-major (row = output column, stride 128).
// One wave computes one 16x16 tile; K=128 -> 4 x v_wmma_f32_16x16x32_bf16.
__global__ void k_gemm_bf16(const __bf16* __restrict__ A, const __bf16* __restrict__ Wt,
                            const float* __restrict__ bias, float* __restrict__ C,
                            int rtiles, int ctiles) {
    int wid = blockIdx.x * (blockDim.x >> 5) + (threadIdx.x >> 5);
    if (wid >= rtiles * ctiles) return;
    int tr = wid / ctiles, tc = wid % ctiles;
    int lane = threadIdx.x & 31;

    const __bf16* ap = A  + (size_t)(tr * 16 + (lane & 15)) * HID;
    const __bf16* bp = Wt + (size_t)(tc * 16 + (lane & 15)) * HID;
    // ISA A layout (16-bit 16x32): lane<16 -> K[0..7],K[16..23]; lane>=16 -> K[8..15],K[24..31]
    int aoff = (lane & 16) ? 8 : 0;
    // ISA B layout (16-bit 32x16): lane<16 -> K[0..7],K[16..23]; lane>=16 -> K[8..15],K[24..31]
    int boff = (lane & 16) ? 8 : 0;

    Acc acc;
    acc.v = (v8f){0.f, 0.f, 0.f, 0.f, 0.f, 0.f, 0.f, 0.f};
#pragma unroll
    for (int kb = 0; kb < 4; ++kb) {
        Frag a, b;
        a.u[0] = *(const uint4*)(ap + kb * 32 + aoff);
        a.u[1] = *(const uint4*)(ap + kb * 32 + aoff + 16);
        b.u[0] = *(const uint4*)(bp + kb * 32 + boff);
        b.u[1] = *(const uint4*)(bp + kb * 32 + boff + 16);
        acc.v = __builtin_amdgcn_wmma_f32_16x16x32_bf16(
            false, a.v, false, b.v, (short)0, acc.v, false, false);
    }
    // C/D layout: VGPR r -> (M = r + (lane<16?0:8), N = lane&15)
    int ccol  = tc * 16 + (lane & 15);
    float bv  = bias ? bias[ccol] : 0.f;
    int rbase = tr * 16 + ((lane & 16) ? 8 : 0);
    int ncols = ctiles * 16;
#pragma unroll
    for (int r = 0; r < 8; ++r)
        C[(size_t)(rbase + r) * ncols + ccol] = acc.f[r] + bv;
}

// ---------- pass 1: gated scores + segment max (1 wave per edge) ----------
__global__ void k_edge1(const float* __restrict__ qkv, const int* __restrict__ ei,
                        const float* __restrict__ dist,
                        const float* __restrict__ g1W, const float* __restrict__ g1b,
                        const float* __restrict__ g2W, const float* __restrict__ g2b,
                        float* __restrict__ scores, unsigned* __restrict__ smaxk) {
    int e = blockIdx.x * (blockDim.x >> 5) + (threadIdx.x >> 5);
    if (e >= E_EDGES) return;
    int lane = threadIdx.x & 31;
    int row = ei[e], col = ei[E_EDGES + e];

    float4 q4 = *(const float4*)(qkv + (size_t)row * QKV_COLS + lane * 4);
    float4 k4 = *(const float4*)(qkv + (size_t)col * QKV_COLS + HID + lane * 4);
    float p = q4.x * k4.x + q4.y * k4.y + q4.z * k4.z + q4.w * k4.w;
    p += __shfl_xor(p, 1);  // reduce within 8-lane head group
    p += __shfl_xor(p, 2);
    p += __shfl_xor(p, 4);

    // gate MLP: 1 lane per hidden unit (GATE == 32 == wave size)
    float dv = dist[e];
    float a  = dv * g1W[lane] + g1b[lane];
    float s  = a / (1.f + __expf(-a));          // SiLU
    float t  = s * g2W[lane];
    t += __shfl_xor(t, 1);  t += __shfl_xor(t, 2);  t += __shfl_xor(t, 4);
    t += __shfl_xor(t, 8);  t += __shfl_xor(t, 16);
    float g = 1.f / (1.f + __expf(-(t + g2b[0])));

    float sc = p * 0.17677669529663687f * g;    // 1/sqrt(32)
    if ((lane & 7) == 0) {
        int hd = lane >> 3;
        scores[(size_t)e * 4 + hd] = sc;
        atomicMax(&smaxk[(size_t)row * 4 + hd], fkey(sc));
    }
}

// ---------- pass 2: exp, denom, and exp*v accumulation ----------
__global__ void k_edge2(const float* __restrict__ qkv, const int* __restrict__ ei,
                        const float* __restrict__ scores, const unsigned* __restrict__ smaxk,
                        float* __restrict__ denom, float* __restrict__ oacc) {
    int e = blockIdx.x * (blockDim.x >> 5) + (threadIdx.x >> 5);
    if (e >= E_EDGES) return;
    int lane = threadIdx.x & 31;
    int row = ei[e], col = ei[E_EDGES + e];
    int hd = lane >> 3;

    float s  = scores[(size_t)e * 4 + hd];
    float m  = funkey(smaxk[(size_t)row * 4 + hd]);
    float ex = __expf(s - m);
    if ((lane & 7) == 0) atomicAdd(&denom[(size_t)row * 4 + hd], ex);

    float4 v4 = *(const float4*)(qkv + (size_t)col * QKV_COLS + 2 * HID + lane * 4);
    float* dst = oacc + (size_t)row * HID + lane * 4;
    atomicAdd(dst + 0, ex * v4.x);
    atomicAdd(dst + 1, ex * v4.y);
    atomicAdd(dst + 2, ex * v4.z);
    atomicAdd(dst + 3, ex * v4.w);
}

// ---------- normalize (sum ex*v / sum ex), cast to bf16 for the final WMMA GEMM ----------
__global__ void k_norm(const float* __restrict__ oacc, const float* __restrict__ denom,
                       __bf16* __restrict__ outb) {
    int i = blockIdx.x * blockDim.x + threadIdx.x;
    if (i >= N_NODES * HID) return;
    int node = i >> 7, hd = (i & 127) >> 5;
    float d = denom[node * 4 + hd];
    outb[i] = (__bf16)((d > 0.f) ? oacc[i] / d : 0.f);  // empty segment -> 0 (final row = ob)
}

extern "C" void kernel_launch(void* const* d_in, const int* in_sizes, int n_in,
                              void* d_out, int out_size, void* d_ws, size_t ws_size,
                              hipStream_t stream) {
    const float* h    = (const float*)d_in[0];
    const float* fc   = (const float*)d_in[1];
    const int*   ei   = (const int*)d_in[2];
    const float* dist = (const float*)d_in[3];
    const float* fW   = (const float*)d_in[4];
    const float* fb   = (const float*)d_in[5];
    const float* qW   = (const float*)d_in[6];
    const float* qb   = (const float*)d_in[7];
    const float* kW   = (const float*)d_in[8];
    const float* kb   = (const float*)d_in[9];
    const float* vW   = (const float*)d_in[10];
    const float* vb   = (const float*)d_in[11];
    const float* oW   = (const float*)d_in[12];
    const float* ob   = (const float*)d_in[13];
    const float* g1W  = (const float*)d_in[14];
    const float* g1b  = (const float*)d_in[15];
    const float* g2W  = (const float*)d_in[16];
    const float* g2b  = (const float*)d_in[17];
    float* out = (float*)d_out;

    char* ws = (char*)d_ws;
    size_t off = 0;
    auto alloc = [&](size_t bytes) -> char* {
        char* p = ws + off;
        off = (off + bytes + 255) & ~(size_t)255;
        return p;
    };
    __bf16*   hcA   = (__bf16*)alloc((size_t)N_NODES * HID * 2);   // reused for normalized attn out
    __bf16*   wqkvT = (__bf16*)alloc((size_t)QKV_COLS * HID * 2);
    float*    bqkv  = (float*)alloc((size_t)QKV_COLS * 4);
    __bf16*   woT   = (__bf16*)alloc((size_t)HID * HID * 2);
    float*    qkv   = (float*)alloc((size_t)N_NODES * QKV_COLS * 4);
    float*    scores= (float*)alloc((size_t)E_EDGES * 4 * 4);
    unsigned* smaxk = (unsigned*)alloc((size_t)N_NODES * 4 * 4);
    float*    denom = (float*)alloc((size_t)N_NODES * 4 * 4);
    float*    oacc  = (float*)alloc((size_t)N_NODES * HID * 4);
    (void)ws_size; (void)n_in; (void)in_sizes; (void)out_size;

    const int TB = 256;
    k_zero<<<(N_NODES * 4 + TB - 1) / TB, TB, 0, stream>>>((float*)smaxk, N_NODES * 4);
    k_zero<<<(N_NODES * 4 + TB - 1) / TB, TB, 0, stream>>>(denom, N_NODES * 4);
    k_zero<<<(N_NODES * HID + TB - 1) / TB, TB, 0, stream>>>(oacc, N_NODES * HID);

    k_hc<<<(N_NODES * HID + TB - 1) / TB, TB, 0, stream>>>(h, fc, fW, fb, hcA);
    k_prep_qkv<<<(QKV_COLS * HID + TB - 1) / TB, TB, 0, stream>>>(qW, qb, kW, kb, vW, vb, wqkvT, bqkv);
    k_prep_o<<<(HID * HID + TB - 1) / TB, TB, 0, stream>>>(oW, woT);

    const int rt = N_NODES / 16;          // 6250 (exact)
    const int ctQ = QKV_COLS / 16;        // 24
    k_gemm_bf16<<<(rt * ctQ + 7) / 8, TB, 0, stream>>>(hcA, wqkvT, bqkv, qkv, rt, ctQ);

    k_edge1<<<(E_EDGES + 7) / 8, TB, 0, stream>>>(qkv, ei, dist, g1W, g1b, g2W, g2b, scores, smaxk);
    k_edge2<<<(E_EDGES + 7) / 8, TB, 0, stream>>>(qkv, ei, scores, smaxk, denom, oacc);

    k_norm<<<(N_NODES * HID + TB - 1) / TB, TB, 0, stream>>>(oacc, denom, hcA);

    const int ctO = HID / 16;             // 8
    k_gemm_bf16<<<(rt * ctO + 7) / 8, TB, 0, stream>>>(hcA, woT, ob, out, rt, ctO);
}
